// QuantizedCpuLinear_44805098832172
// MI455X (gfx1250) — matrix operand order repaired and checked by reading, not compiled, not run on hardware
//
#include <hip/hip_runtime.h>

// QuantizedLinear: out = x @ (w_int8 * scale)^T
//   M = B*S = 8192, K = IN = 4096, N = OUT = 11008  (fp32 in/out)
// Compute-bound (AI ~1100 FLOP/B) -> v_wmma_f32_16x16x32_f16.
// int8 weights exact in f16; scalar scale applied in f32 epilogue.
// Ping-pong LDS double buffering + BK=64 => 16 WMMA per single barrier.

#define M_TOTAL 8192
#define N_TOTAL 11008
#define K_TOTAL 4096

constexpr int BM  = 128;  // block tile M
constexpr int BN  = 128;  // block tile N
constexpr int BK  = 64;   // block tile K = 2 WMMA K-substeps
constexpr int LDT = 72;   // padded LDS row stride in halves (144B rows: 16B-aligned frags)

typedef __attribute__((ext_vector_type(16))) _Float16 v16h;
typedef __attribute__((ext_vector_type(8)))  _Float16 v8h;
typedef __attribute__((ext_vector_type(4)))  _Float16 v4h;
typedef __attribute__((ext_vector_type(8)))  float    v8f;

__global__ __launch_bounds__(256)
void qlinear_wmma_f16(const float* __restrict__ x,
                      const int*   __restrict__ w,
                      const float* __restrict__ scale_p,
                      float*       __restrict__ out)
{
    // Ping-pong buffers: write tile kt+1 into buf[1-p] while computing buf[p].
    __shared__ _Float16 Alds[2][BM * LDT];
    __shared__ _Float16 Blds[2][BN * LDT];

    const int tid   = threadIdx.x;
    const int lane  = tid & 31;
    const int wave  = tid >> 5;          // 8 waves
    const int waveM = wave & 3;          // 4 waves along M -> 32 rows each
    const int waveN = wave >> 2;         // 2 waves along N -> 64 cols each
    const int m0    = blockIdx.y * BM;
    const int n0    = blockIdx.x * BN;

    // Staging: 256 threads cover 128 rows x 64 cols per tile; each thread
    // 8x float4/int4 (rows r0+16i, cols c4..c4+3). 16 threads span a 256B row.
    const int c4 = (tid & 15) * 4;       // k-offset within tile (0..60)
    const int r0 = tid >> 4;             // 0..15

    const float* xg = x + (size_t)m0 * K_TOTAL + c4;
    const int*   wg = w + (size_t)n0 * K_TOTAL + c4;

    v8f acc[2][4] = {};                  // 8 accumulators: 2 M-sub x 4 N-sub
    float4 aS[8];
    int4   bS[8];

    auto load_stage = [&](int kt) {
        const float* xa = xg + kt * BK;
        const int*   wa = wg + kt * BK;
#pragma unroll
        for (int i = 0; i < 8; ++i) {
            const int row = r0 + 16 * i;
            aS[i] = *(const float4*)(xa + (size_t)row * K_TOTAL);
            bS[i] = *(const int4*)  (wa + (size_t)row * K_TOTAL);
        }
    };

    auto store_stage = [&](int buf) {
#pragma unroll
        for (int i = 0; i < 8; ++i) {
            const int row = r0 + 16 * i;
            v4h av = { (_Float16)aS[i].x, (_Float16)aS[i].y,
                       (_Float16)aS[i].z, (_Float16)aS[i].w };
            *(v4h*)&Alds[buf][row * LDT + c4] = av;      // 8B ds_store
            v4h bv = { (_Float16)(float)bS[i].x, (_Float16)(float)bS[i].y,
                       (_Float16)(float)bS[i].z, (_Float16)(float)bS[i].w };
            *(v4h*)&Blds[buf][row * LDT + c4] = bv;      // int8 exact in f16
        }
    };

    constexpr int KT = K_TOTAL / BK;     // 64 staged K-tiles

    load_stage(0);
    store_stage(0);
    __syncthreads();

    const int kh8  = (lane >> 4) * 8;    // A frag K base per lane-half
    const int kh16 = (lane >> 4) * 16;   // unused: B frag also uses kh8
    const int lr   = lane & 15;

    for (int kt = 0; kt < KT; ++kt) {
        const int p = kt & 1;
        if (kt + 1 < KT) load_stage(kt + 1);              // overlap with WMMA
        if (kt + 2 < KT) {                                // prime L2 (global_prefetch_b8)
            __builtin_prefetch(xg + (size_t)r0 * K_TOTAL + (size_t)(kt + 2) * BK, 0, 1);
            __builtin_prefetch(wg + (size_t)r0 * K_TOTAL + (size_t)(kt + 2) * BK, 0, 1);
        }

#pragma unroll
        for (int ks = 0; ks < 2; ++ks) {                  // 2 WMMA K-substeps per tile
            const int ko = ks * 32;
            // A frag (16-bit A 16x32): lanes 0-15 -> K{0-7,16-23},
            // lanes 16-31 -> K{8-15,24-31}; M = lane&15.
            v16h af[2];
#pragma unroll
            for (int mi = 0; mi < 2; ++mi) {
                const int row = waveM * 32 + mi * 16 + lr;
                v8h lo = *(const v8h*)&Alds[p][row * LDT + ko + kh8];
                v8h hi = *(const v8h*)&Alds[p][row * LDT + ko + kh8 + 16];
                af[mi] = __builtin_shufflevector(lo, hi,
                         0,1,2,3,4,5,6,7,8,9,10,11,12,13,14,15);
            }
            // B frag (16-bit B 32x16): lanes 0-15 -> K{0-7,16-23},
            // lanes 16-31 -> K{8-15,24-31}; N = lane&15.
            v16h bf[4];
#pragma unroll
            for (int ni = 0; ni < 4; ++ni) {
                const int row = waveN * 64 + ni * 16 + lr;
                v8h lo = *(const v8h*)&Blds[p][row * LDT + ko + kh8];
                v8h hi = *(const v8h*)&Blds[p][row * LDT + ko + kh8 + 16];
                bf[ni] = __builtin_shufflevector(lo, hi,
                         0,1,2,3,4,5,6,7,8,9,10,11,12,13,14,15);
            }
#pragma unroll
            for (int mi = 0; mi < 2; ++mi)
#pragma unroll
                for (int ni = 0; ni < 4; ++ni)
                    acc[mi][ni] = __builtin_amdgcn_wmma_f32_16x16x32_f16(
                        /*neg_a=*/false, af[mi], /*neg_b=*/false, bf[ni],
                        /*c_mod=*/(short)0, acc[mi][ni],
                        /*reuse_a=*/false, /*reuse_b=*/false);
        }

        if (kt + 1 < KT) store_stage(1 - p);  // publish next tile to other buffer
        __syncthreads();                      // single barrier per K-tile
    }

    // Epilogue: D 16x16 f32 layout: lane<16 -> N=lane, M=r; lane>=16 -> N=lane-16, M=8+r.
    const float s = scale_p[0];
    const int colBase = n0 + waveN * 64 + lr;
    const int rowBase = m0 + waveM * 32 + ((lane >> 4) * 8);
#pragma unroll
    for (int mi = 0; mi < 2; ++mi) {
#pragma unroll
        for (int ni = 0; ni < 4; ++ni) {
            const int col = colBase + ni * 16;
#pragma unroll
            for (int r = 0; r < 8; ++r) {
                const int row = rowBase + mi * 16 + r;
                out[(size_t)row * N_TOTAL + col] = s * acc[mi][ni][r];
            }
        }
    }
}

extern "C" void kernel_launch(void* const* d_in, const int* in_sizes, int n_in,
                              void* d_out, int out_size, void* d_ws, size_t ws_size,
                              hipStream_t stream) {
    const float* x     = (const float*)d_in[0];   // [B,S,IN] fp32
    const int*   w     = (const int*)  d_in[1];   // [OUT,IN] int32 (int8 range)
    const float* scale = (const float*)d_in[2];   // [1] fp32
    float*       out   = (float*)d_out;           // [B,S,OUT] fp32

    dim3 grid(N_TOTAL / BN, M_TOTAL / BM);        // (86, 64)
    qlinear_wmma_f16<<<grid, 256, 0, stream>>>(x, w, scale, out);
}
